// GNNModel_2173253451837
// MI455X (gfx1250) — matrix operand order repaired and checked by reading, not compiled, not run on hardware
//
#include <hip/hip_runtime.h>
#include <hip/hip_bf16.h>

// ---------------------------------------------------------------------------
// MI455X (gfx1250, wave32) implementation of the GATv2 GNN forward pass.
// Dense GEMMs use v_wmma_f32_16x16x32_f16 (f16 inputs, f32 accumulate) with
// double-buffered LDS and a register-level software pipeline (global loads for
// tile i+1 issued before computing tile i); sparse graph ops use wave-per-edge
// VALU kernels with f32 global atomics.
// ---------------------------------------------------------------------------

#define NNODE   50000
#define NEDGE   250000
#define NAUG    (NEDGE + NNODE)   // edges + self loops
#define INDIM   128
#define OUTDIM  256
#define EDGEDIM 64
#define HEADS   4
#define D1      256
#define D2      256
#define CATOUT  512
#define LNEPS   1e-5f

typedef __attribute__((ext_vector_type(16))) _Float16 v16h;
typedef __attribute__((ext_vector_type(8)))  _Float16 v8h;
typedef __attribute__((ext_vector_type(4)))  _Float16 h4;
typedef __attribute__((ext_vector_type(8)))  float    v8f;
typedef __attribute__((ext_vector_type(4)))  float    f32x4;

// ------------------------- float <-> ordered uint (for f32 atomicMax) ------
__device__ __forceinline__ unsigned f2ord(float f) {
    unsigned u = __float_as_uint(f);
    return (u & 0x80000000u) ? ~u : (u | 0x80000000u);
}
__device__ __forceinline__ float ord2f(unsigned u) {
    return __uint_as_float((u & 0x80000000u) ? (u & 0x7fffffffu) : ~u);
}

// ===========================================================================
// Tiled WMMA GEMM:  C[M,N] = act(A[M,K] @ B[K,N] + bias)
// Block = 256 threads = 8 waves; block tile 32(M) x 128(N); K step 32.
// Each wave owns two 16x16 tiles (shared A fragment, 2 WMMAs per k-step).
// Software pipeline: tile i+1 global loads are issued (into registers) right
// after the barrier, while WMMAs for tile i run; they are converted and
// committed to the other LDS buffer at the top of the next iteration.
// A stored row-major, B transposed in LDS so fragment fills are ds_load_b128
// pairs.  OOB handled by index clamping on loads + guards on stores.
// ACT: 0 = none, 1 = relu, 2 = elu
// ===========================================================================
#define ALD 40   // padded LDS row stride in halves (80 B: 16-B aligned, bank-spread)

template <int ACT>
__global__ __launch_bounds__(256) void gemm_wmma(
    const float* __restrict__ A, const float* __restrict__ B,
    const float* __restrict__ bias, float* __restrict__ C,
    int M, int N, int K)
{
    __shared__ _Float16 As[2][32 * ALD];    //  2 x 2.5 KB
    __shared__ _Float16 Bt[2][128 * ALD];   //  2 x 10 KB  (transposed: [n][k])

    const int tid  = threadIdx.x;
    const int lane = tid & 31;
    const int wave = tid >> 5;
    const int wr   = wave >> 2;          // 0..1 : 16-row slice
    const int wc   = wave & 3;           // 0..3 : 32-col slice (two 16-col tiles)
    const int hp   = lane >> 4;          // wave32 lane half
    const int l15  = lane & 15;

    const long long m0 = (long long)blockIdx.x * 32;
    const int       n0 = blockIdx.y * 128;

    // ---- staging geometry (clamped indices: unconditional vector loads) ----
    const int arow = tid >> 3;                       // 0..31
    const int ac4  = (tid & 7) * 4;                  // 0,4,..,28
    long long agm  = m0 + arow; if (agm > M - 1) agm = M - 1;
    const int asto = arow * ALD + ac4;               // A LDS store offset

    const size_t sN = (size_t)N;
    const float* ap = A + agm * (long long)K + ac4;  // bumped by 32 per k-step
    const float* bp[4];
    int bsto[4];
    #pragma unroll
    for (int u = 0; u < 4; ++u) {
        int idx = tid + u * 256;
        int nl  = idx & 127;
        int k4  = (idx >> 7) * 4;
        int gn  = n0 + nl; if (gn > N - 1) gn = N - 1;
        bp[u]   = B + (size_t)k4 * sN + gn;          // bumped by 32*N per k-step
        bsto[u] = nl * ALD + k4;
    }

    // fragment LDS offsets (halves)
    const int aoff  = (wr * 16 + l15) * ALD;
    const int boff0 = (wc * 32 + l15) * ALD;
    const int boff1 = boff0 + 16 * ALD;

    v8f acc0 = {}, acc1 = {};
    const int nk = K >> 5;

    f32x4 areg;          // staged A values (tile in flight)
    f32x4 breg[4];       // staged B values

    auto load_regs = [&]() {               // issue global loads (no LDS yet)
        areg = *(const f32x4*)ap; ap += 32;
        #pragma unroll
        for (int u = 0; u < 4; ++u) {
            breg[u][0] = bp[u][0];
            breg[u][1] = bp[u][sN];
            breg[u][2] = bp[u][2 * sN];
            breg[u][3] = bp[u][3 * sN];
            bp[u] += 32 * sN;
        }
    };
    auto commit_lds = [&](int buf) {       // convert f32->f16, write LDS
        h4 ah;
        #pragma unroll
        for (int j = 0; j < 4; ++j) ah[j] = (_Float16)areg[j];
        *(h4*)&As[buf][asto] = ah;
        #pragma unroll
        for (int u = 0; u < 4; ++u) {
            h4 bh;
            #pragma unroll
            for (int j = 0; j < 4; ++j) bh[j] = (_Float16)breg[u][j];
            *(h4*)&Bt[buf][bsto[u]] = bh;
        }
    };

    load_regs();                           // prologue: tile 0 in registers
    int buf = 0;
    for (int i = 0; i < nk; ++i) {
        commit_lds(buf);                   // waits tile-i loads, fills LDS[buf]
        __syncthreads();
        if (i + 1 < nk) load_regs();       // tile i+1 loads overlap WMMAs below

        const _Float16* Ab = As[buf];
        const _Float16* Bb = Bt[buf];

        // A fragment (16-bit A 16x32 layout): two contiguous 8-half runs:
        // k = hp*8..+7 and k = 16+hp*8..+7  -> two ds_load_b128
        const v8h* Arow = (const v8h*)&Ab[aoff];
        v16h af = __builtin_shufflevector(Arow[hp], Arow[2 + hp],
                      0,1,2,3,4,5,6,7,8,9,10,11,12,13,14,15);

        // B fragments: same k order as A (k = hp*8..+7 and 16+hp*8..+7)
        const v8h* Bc0 = (const v8h*)&Bb[boff0];
        v16h bf0 = __builtin_shufflevector(Bc0[hp], Bc0[2 + hp],
                      0,1,2,3,4,5,6,7,8,9,10,11,12,13,14,15);
        const v8h* Bc1 = (const v8h*)&Bb[boff1];
        v16h bf1 = __builtin_shufflevector(Bc1[hp], Bc1[2 + hp],
                      0,1,2,3,4,5,6,7,8,9,10,11,12,13,14,15);

        acc0 = __builtin_amdgcn_wmma_f32_16x16x32_f16(
                   false, af, false, bf0, (short)0, acc0, false, false);
        acc1 = __builtin_amdgcn_wmma_f32_16x16x32_f16(
                   false, af, false, bf1, (short)0, acc1, false, false);
        buf ^= 1;
    }

    // ---- store: f32 C/D layout: VGPR r -> M = r + 8*half, N = lane&15 -----
    #pragma unroll
    for (int t = 0; t < 2; ++t) {
        const v8f& a = t ? acc1 : acc0;
        int gn = n0 + wc * 32 + t * 16 + l15;
        if (gn >= N) continue;
        float bv = bias ? bias[gn] : 0.0f;
        #pragma unroll
        for (int r = 0; r < 8; ++r) {
            long long gm = m0 + wr * 16 + hp * 8 + r;
            if (gm < M) {
                float v = a[r] + bv;
                if (ACT == 1)      v = fmaxf(v, 0.0f);
                else if (ACT == 2) v = (v > 0.0f) ? v : (__expf(v) - 1.0f);
                C[gm * (long long)N + gn] = v;
            }
        }
    }
}

// ===========================================================================
// BatchNorm over nodes (column statistics for 128 features)
// ===========================================================================
__global__ void bn_colstats(const float* __restrict__ x,
                            float* __restrict__ gsum, float* __restrict__ gsq,
                            int rows)
{
    int c = threadIdx.x;                         // blockDim = 128 (one col/lane)
    float s = 0.0f, q = 0.0f;
    for (long long i = blockIdx.x; i < rows; i += gridDim.x) {
        float v = x[i * INDIM + c];
        s += v; q += v * v;
    }
    atomicAdd(&gsum[c], s);
    atomicAdd(&gsq[c], q);
}

__global__ void bn_apply(const float* __restrict__ x,
                         const float* __restrict__ gsum,
                         const float* __restrict__ gsq,
                         const float* __restrict__ g,
                         const float* __restrict__ b,
                         float* __restrict__ y, long long n)
{
    long long idx = (long long)blockIdx.x * blockDim.x + threadIdx.x;
    if (idx >= n) return;
    int c = (int)(idx & (INDIM - 1));
    const float invN = 1.0f / (float)NNODE;
    float mu  = gsum[c] * invN;
    float var = gsq[c] * invN - mu * mu;
    y[idx] = (x[idx] - mu) * rsqrtf(var + LNEPS) * g[c] + b[c];
}

// ===========================================================================
// Row LayerNorm (+ optional pre-bias and leaky-relu).  One wave per row.
// ===========================================================================
template <int COLS>
__global__ void row_ln_leaky(const float* __restrict__ X,
                             const float* __restrict__ prebias,
                             const float* __restrict__ g,
                             const float* __restrict__ b,
                             float slope, long long rows,
                             float* __restrict__ Y)
{
    constexpr int PER = COLS / 32;
    int lane = threadIdx.x & 31;
    long long row = (long long)blockIdx.x * (blockDim.x >> 5) + (threadIdx.x >> 5);
    if (row >= rows) return;
    const float* xr = X + row * COLS;
    float v[PER];
    float s = 0.0f;
    #pragma unroll
    for (int i = 0; i < PER; ++i) {
        int c = lane * PER + i;
        float t = xr[c];
        if (prebias) t += prebias[c];
        v[i] = t; s += t;
    }
    #pragma unroll
    for (int off = 16; off; off >>= 1) s += __shfl_xor(s, off, 32);
    float mu = s * (1.0f / COLS);
    float q = 0.0f;
    #pragma unroll
    for (int i = 0; i < PER; ++i) { float d = v[i] - mu; q += d * d; }
    #pragma unroll
    for (int off = 16; off; off >>= 1) q += __shfl_xor(q, off, 32);
    float r = rsqrtf(q * (1.0f / COLS) + LNEPS);
    #pragma unroll
    for (int i = 0; i < PER; ++i) {
        int c = lane * PER + i;
        float t = (v[i] - mu) * r * g[c] + b[c];
        Y[row * COLS + c] = (t > 0.0f) ? t : slope * t;
    }
}

// ===========================================================================
// Edge classifier second layer (GEMV 32->1) + sigmoid gate
// ===========================================================================
__global__ void ec2_sigmoid(const float* __restrict__ h,
                            const float* __restrict__ W2,
                            const float* __restrict__ b2,
                            float* __restrict__ w, long long E_)
{
    long long e = (long long)blockIdx.x * blockDim.x + threadIdx.x;
    if (e >= E_) return;
    float acc = b2[0];
    const float* hr = h + e * 32;
    #pragma unroll
    for (int j = 0; j < 32; ++j) acc += hr[j] * W2[j];
    w[e] = 1.0f / (1.0f + __expf(-acc));
}

// ===========================================================================
// Gate encoded edge features in-place and scatter segment sums / counts.
// One wave per edge; 8 contiguous f32 channels per lane (256 total).
// ===========================================================================
__global__ void gate_scatter(float* __restrict__ encA,
                             const float* __restrict__ wgate,
                             const int* __restrict__ ei,
                             float* __restrict__ seg_sum,
                             float* __restrict__ cnt, long long E_)
{
    int lane = threadIdx.x & 31;
    long long e = (long long)blockIdx.x * (blockDim.x >> 5) + (threadIdx.x >> 5);
    if (e >= E_) return;
    int d = ei[E_ + e];
    float wv = wgate[e];
    float* row = encA + e * OUTDIM + lane * 8;
    float* seg = seg_sum + (size_t)d * OUTDIM + lane * 8;
    f32x4 r0 = *(f32x4*)row, r1 = *(f32x4*)(row + 4);
    #pragma unroll
    for (int j = 0; j < 4; ++j) { r0[j] *= wv; r1[j] *= wv; }
    *(f32x4*)row = r0; *(f32x4*)(row + 4) = r1;
    #pragma unroll
    for (int j = 0; j < 4; ++j) {
        atomicAdd(&seg[j],     r0[j]);
        atomicAdd(&seg[4 + j], r1[j]);
    }
    if (lane == 0) atomicAdd(&cnt[d], 1.0f);
}

// loop_attr (== segment mean) written into encA rows [E, E+N)
__global__ void loop_attr_fill(const float* __restrict__ seg_sum,
                               const float* __restrict__ cnt,
                               float* __restrict__ encA, long long n)
{
    long long idx = (long long)blockIdx.x * blockDim.x + threadIdx.x;
    if (idx >= n) return;
    long long i = idx >> 8;
    float ct = fmaxf(cnt[i], 1.0f);
    encA[(size_t)(NEDGE + i) * OUTDIM + (idx & 255)] = seg_sum[idx] / ct;
}

// ===========================================================================
// GATv2 attention.  One wave per augmented edge; lane owns 8 channels,
// 8-lane groups map to heads (head = lane>>3).
// ===========================================================================
__global__ void att_logits_max(const float* __restrict__ xl,
                               const float* __restrict__ xr,
                               const float* __restrict__ eeg,
                               const float* __restrict__ att,
                               const int* __restrict__ ei,
                               float* __restrict__ logit_out,
                               unsigned* __restrict__ m_u)
{
    int lane = threadIdx.x & 31;
    long long e = (long long)blockIdx.x * (blockDim.x >> 5) + (threadIdx.x >> 5);
    if (e >= NAUG) return;
    int s, d;
    if (e < NEDGE) { s = ei[e]; d = ei[NEDGE + e]; }
    else           { s = d = (int)(e - NEDGE); }
    const float* xls = xl  + (size_t)s * D1 + lane * 8;
    const float* xrd = xr  + (size_t)d * D1 + lane * 8;
    const float* ee  = eeg + (size_t)e * D1 + lane * 8;
    const float* at  = att + lane * 8;
    f32x4 a0 = *(const f32x4*)xls, a1 = *(const f32x4*)(xls + 4);
    f32x4 b0 = *(const f32x4*)xrd, b1 = *(const f32x4*)(xrd + 4);
    f32x4 c0 = *(const f32x4*)ee,  c1 = *(const f32x4*)(ee + 4);
    f32x4 t0 = *(const f32x4*)at,  t1 = *(const f32x4*)(at + 4);
    float p = 0.0f;
    #pragma unroll
    for (int j = 0; j < 4; ++j) {
        float z0 = a0[j] + b0[j] + c0[j];
        float z1 = a1[j] + b1[j] + c1[j];
        z0 = (z0 > 0.0f) ? z0 : 0.2f * z0;    // leaky 0.2
        z1 = (z1 > 0.0f) ? z1 : 0.2f * z1;
        p += z0 * t0[j] + z1 * t1[j];
    }
    p += __shfl_xor(p, 1, 32);
    p += __shfl_xor(p, 2, 32);
    p += __shfl_xor(p, 4, 32);
    if ((lane & 7) == 0) {
        int h = lane >> 3;
        logit_out[e * HEADS + h] = p;
        atomicMax(&m_u[(size_t)d * HEADS + h], f2ord(p));
    }
}

__global__ void att_exp(float* __restrict__ logit_ex,
                        const unsigned* __restrict__ m_u,
                        float* __restrict__ den,
                        const int* __restrict__ ei)
{
    long long idx = (long long)blockIdx.x * blockDim.x + threadIdx.x;
    if (idx >= (long long)NAUG * HEADS) return;
    long long e = idx >> 2; int h = (int)(idx & 3);
    int d = (e < NEDGE) ? ei[NEDGE + e] : (int)(e - NEDGE);
    float m  = ord2f(m_u[(size_t)d * HEADS + h]);
    float ex = __expf(logit_ex[idx] - m);
    logit_ex[idx] = ex;
    atomicAdd(&den[(size_t)d * HEADS + h], ex);
}

__global__ void att_norm_agg(const float* __restrict__ xl,
                             float* __restrict__ ex_alpha,
                             const float* __restrict__ den,
                             const int* __restrict__ ei,
                             float* __restrict__ outacc)
{
    int lane = threadIdx.x & 31;
    long long e = (long long)blockIdx.x * (blockDim.x >> 5) + (threadIdx.x >> 5);
    if (e >= NAUG) return;
    int s, d;
    if (e < NEDGE) { s = ei[e]; d = ei[NEDGE + e]; }
    else           { s = d = (int)(e - NEDGE); }
    int h = lane >> 3;
    float a = ex_alpha[e * HEADS + h] / den[(size_t)d * HEADS + h];
    if ((lane & 7) == 0) ex_alpha[e * HEADS + h] = a;   // final alpha output
    const float* xls = xl + (size_t)s * D1 + lane * 8;
    float* od = outacc + (size_t)d * D1 + lane * 8;
    f32x4 v0 = *(const f32x4*)xls, v1 = *(const f32x4*)(xls + 4);
    #pragma unroll
    for (int j = 0; j < 4; ++j) {
        atomicAdd(&od[j],     v0[j] * a);
        atomicAdd(&od[4 + j], v1[j] * a);
    }
}

// ===========================================================================
// Small elementwise kernels
// ===========================================================================
__global__ void axpby(const float* __restrict__ X, const float* __restrict__ Y,
                      float a, float* __restrict__ Z, long long n)
{
    long long idx = (long long)blockIdx.x * blockDim.x + threadIdx.x;
    if (idx < n) Z[idx] = a * X[idx] + Y[idx];
}

__global__ void fuse_concat(const float* __restrict__ fx1,
                            const float* __restrict__ x2,
                            const float* __restrict__ efm,
                            const float* __restrict__ alpha,
                            float* __restrict__ xf, long long n)
{
    long long idx = (long long)blockIdx.x * blockDim.x + threadIdx.x;
    if (idx >= n) return;
    long long i = idx >> 9;          // row (CATOUT=512)
    int c = (int)(idx & 511);
    float s = 1.0f / (1.0f + __expf(-alpha[0]));
    xf[idx] = (c < 256) ? s * fx1[i * 256 + c] + (1.0f - s) * x2[i * 256 + c]
                        : efm[i * 256 + (c - 256)];
}

__global__ void np3_gemv(const float* __restrict__ h2,
                         const float* __restrict__ W3,
                         const float* __restrict__ b3,
                         float* __restrict__ out, long long n)
{
    long long i = (long long)blockIdx.x * blockDim.x + threadIdx.x;
    if (i >= n) return;
    float acc = b3[0];
    const float* hr = h2 + i * 128;
    #pragma unroll
    for (int j = 0; j < 128; ++j) acc += hr[j] * W3[j];
    out[i] = acc;
}

// ===========================================================================
// Host-side orchestration
// ===========================================================================
extern "C" void kernel_launch(void* const* d_in, const int* in_sizes, int n_in,
                              void* d_out, int out_size, void* d_ws, size_t ws_size,
                              hipStream_t stream)
{
    (void)in_sizes; (void)n_in; (void)out_size; (void)ws_size;

    // ---- inputs (setup_inputs dict order, params flattened in order) -------
    const float* x_in   = (const float*)d_in[0];
    const int*   ei     = (const int*)  d_in[1];   // [2,E] src row then dst row
    const float* eattr  = (const float*)d_in[2];
    const float* bn0_g  = (const float*)d_in[3];
    const float* bn0_b  = (const float*)d_in[4];
    const float* proj_W = (const float*)d_in[5];
    const float* proj_b = (const float*)d_in[6];
    const float* eeln_g = (const float*)d_in[7];
    const float* eeln_b = (const float*)d_in[8];
    const float* ee_W1  = (const float*)d_in[9];
    const float* ee_b1  = (const float*)d_in[10];
    const float* ee_W2  = (const float*)d_in[11];
    const float* ee_b2  = (const float*)d_in[12];
    const float* ec_W1  = (const float*)d_in[13];
    const float* ec_b1  = (const float*)d_in[14];
    const float* ec_W2  = (const float*)d_in[15];
    const float* ec_b2  = (const float*)d_in[16];
    const float* g1_Wl  = (const float*)d_in[17];
    const float* g1_bl  = (const float*)d_in[18];
    const float* g1_Wr  = (const float*)d_in[19];
    const float* g1_br  = (const float*)d_in[20];
    const float* g1_We  = (const float*)d_in[21];
    const float* g1_att = (const float*)d_in[22];
    const float* g1_bias= (const float*)d_in[23];
    const float* g2_Wl  = (const float*)d_in[24];
    const float* g2_bl  = (const float*)d_in[25];
    const float* g2_Wr  = (const float*)d_in[26];
    const float* g2_br  = (const float*)d_in[27];
    const float* g2_We  = (const float*)d_in[28];
    const float* g2_att = (const float*)d_in[29];
    const float* g2_bias= (const float*)d_in[30];
    const float* ln1_g  = (const float*)d_in[31];
    const float* ln1_b  = (const float*)d_in[32];
    const float* ln2_g  = (const float*)d_in[33];
    const float* ln2_b  = (const float*)d_in[34];
    const float* fp_W   = (const float*)d_in[35];
    const float* fp_b   = (const float*)d_in[36];
    const float* alpha  = (const float*)d_in[37];
    const float* np_W1  = (const float*)d_in[38];
    const float* np_b1  = (const float*)d_in[39];
    const float* np_W2  = (const float*)d_in[40];
    const float* np_b2  = (const float*)d_in[41];
    const float* np_W3  = (const float*)d_in[42];
    const float* np_b3  = (const float*)d_in[43];

    // ---- output layout: xf | node_probs | attn1 | attn2 --------------------
    float* out      = (float*)d_out;
    float* xf_out   = out;                                   // N*512
    float* npb_out  = out + (size_t)NNODE * CATOUT;          // N
    float* attn1    = npb_out + NNODE;                       // NAUG*4
    float* attn2    = attn1 + (size_t)NAUG * HEADS;          // NAUG*4

    // ---- workspace bump allocator ------------------------------------------
    char* wp = (char*)d_ws;
    auto alloc = [&](size_t nfloats) -> float* {
        float* p = (float*)wp;
        wp += ((nfloats * sizeof(float) + 255) / 256) * 256;
        return p;
    };
    float* gsum    = alloc(INDIM);
    float* gsq     = alloc(INDIM);
    float* x_norm  = alloc((size_t)NNODE * INDIM);
    float* e_ln    = alloc((size_t)NEDGE * EDGEDIM);
    float* h_e     = alloc((size_t)NEDGE * 128);
    float* ec_h    = alloc((size_t)NEDGE * 32);
    float* w_gate  = alloc(NEDGE);
    float* encA    = alloc((size_t)NAUG * OUTDIM);
    float* seg_sum = alloc((size_t)NNODE * OUTDIM);
    float* cnt     = alloc(NNODE);
    float* ee_g    = alloc((size_t)NAUG * D1);
    float* xl      = alloc((size_t)NNODE * D1);
    float* xr      = alloc((size_t)NNODE * D1);
    unsigned* m_u  = (unsigned*)alloc((size_t)NNODE * HEADS);
    float* den     = alloc((size_t)NNODE * HEADS);
    float* x1      = alloc((size_t)NNODE * D1);
    float* projx   = alloc((size_t)NNODE * D1);
    float* skip1   = alloc((size_t)NNODE * D1);
    float* x2      = alloc((size_t)NNODE * D2);
    float* fx1     = alloc((size_t)NNODE * D2);
    float* h1      = alloc((size_t)NNODE * 256);
    float* h2      = alloc((size_t)NNODE * 128);

    auto gemm = [&](const float* A, const float* B, const float* bias, float* C,
                    int M, int Nn, int K, int act) {
        dim3 grid((M + 31) / 32, (Nn + 127) / 128);
        if (act == 0)      gemm_wmma<0><<<grid, 256, 0, stream>>>(A, B, bias, C, M, Nn, K);
        else if (act == 1) gemm_wmma<1><<<grid, 256, 0, stream>>>(A, B, bias, C, M, Nn, K);
        else               gemm_wmma<2><<<grid, 256, 0, stream>>>(A, B, bias, C, M, Nn, K);
    };
    auto cdiv = [](long long a, long long b) { return (unsigned)((a + b - 1) / b); };

    // ---- 1. BatchNorm over nodes ------------------------------------------
    hipMemsetAsync(gsum, 0, INDIM * sizeof(float), stream);
    hipMemsetAsync(gsq,  0, INDIM * sizeof(float), stream);
    bn_colstats<<<1024, 128, 0, stream>>>(x_in, gsum, gsq, NNODE);
    bn_apply<<<cdiv((long long)NNODE * INDIM, 256), 256, 0, stream>>>(
        x_in, gsum, gsq, bn0_g, bn0_b, x_norm, (long long)NNODE * INDIM);

    // ---- 2. edge encoder + gate -------------------------------------------
    row_ln_leaky<EDGEDIM><<<cdiv(NEDGE, 8), 256, 0, stream>>>(
        eattr, nullptr, eeln_g, eeln_b, 1.0f, NEDGE, e_ln);
    gemm(e_ln, ee_W1, ee_b1, h_e, NEDGE, 128, EDGEDIM, 1);        // relu
    gemm(h_e, ee_W2, ee_b2, encA, NEDGE, OUTDIM, 128, 0);         // -> encA rows [0,E)
    gemm(eattr, ec_W1, ec_b1, ec_h, NEDGE, 32, EDGEDIM, 1);       // relu
    ec2_sigmoid<<<cdiv(NEDGE, 256), 256, 0, stream>>>(ec_h, ec_W2, ec_b2, w_gate, NEDGE);

    hipMemsetAsync(seg_sum, 0, (size_t)NNODE * OUTDIM * sizeof(float), stream);
    hipMemsetAsync(cnt,     0, NNODE * sizeof(float), stream);
    gate_scatter<<<cdiv(NEDGE, 8), 256, 0, stream>>>(encA, w_gate, ei, seg_sum, cnt, NEDGE);
    loop_attr_fill<<<cdiv((long long)NNODE * OUTDIM, 256), 256, 0, stream>>>(
        seg_sum, cnt, encA, (long long)NNODE * OUTDIM);

    // ---- 3. GAT layer 1 ----------------------------------------------------
    gemm(x_norm, g1_Wl, g1_bl, xl, NNODE, D1, INDIM, 0);
    gemm(x_norm, g1_Wr, g1_br, xr, NNODE, D1, INDIM, 0);
    gemm(encA, g1_We, nullptr, ee_g, NAUG, D1, OUTDIM, 0);

    hipMemsetAsync(m_u, 0, (size_t)NNODE * HEADS * sizeof(unsigned), stream);
    hipMemsetAsync(den, 0, (size_t)NNODE * HEADS * sizeof(float), stream);
    hipMemsetAsync(x1,  0, (size_t)NNODE * D1 * sizeof(float), stream);
    att_logits_max<<<cdiv(NAUG, 8), 256, 0, stream>>>(xl, xr, ee_g, g1_att, ei, attn1, m_u);
    att_exp<<<cdiv((long long)NAUG * HEADS, 256), 256, 0, stream>>>(attn1, m_u, den, ei);
    att_norm_agg<<<cdiv(NAUG, 8), 256, 0, stream>>>(xl, attn1, den, ei, x1);
    row_ln_leaky<D1><<<cdiv(NNODE, 8), 256, 0, stream>>>(
        x1, g1_bias, ln1_g, ln1_b, 0.01f, NNODE, x1);   // in-place: bias+LN+leaky

    gemm(x_norm, proj_W, proj_b, projx, NNODE, D1, INDIM, 0);
    axpby<<<cdiv((long long)NNODE * D1, 256), 256, 0, stream>>>(
        projx, x1, 0.01f, skip1, (long long)NNODE * D1);

    // ---- 4. GAT layer 2 ----------------------------------------------------
    gemm(skip1, g2_Wl, g2_bl, xl, NNODE, D2, D1, 0);
    gemm(skip1, g2_Wr, g2_br, xr, NNODE, D2, D1, 0);
    gemm(encA, g2_We, nullptr, ee_g, NAUG, D2, OUTDIM, 0);

    hipMemsetAsync(m_u, 0, (size_t)NNODE * HEADS * sizeof(unsigned), stream);
    hipMemsetAsync(den, 0, (size_t)NNODE * HEADS * sizeof(float), stream);
    hipMemsetAsync(x2,  0, (size_t)NNODE * D2 * sizeof(float), stream);
    att_logits_max<<<cdiv(NAUG, 8), 256, 0, stream>>>(xl, xr, ee_g, g2_att, ei, attn2, m_u);
    att_exp<<<cdiv((long long)NAUG * HEADS, 256), 256, 0, stream>>>(attn2, m_u, den, ei);
    att_norm_agg<<<cdiv(NAUG, 8), 256, 0, stream>>>(xl, attn2, den, ei, x2);
    row_ln_leaky<D2><<<cdiv(NNODE, 8), 256, 0, stream>>>(
        x2, g2_bias, ln2_g, ln2_b, 0.01f, NNODE, x2);

    // ---- 5. fusion + concat -----------------------------------------------
    gemm(x1, fp_W, fp_b, fx1, NNODE, D2, D1, 0);
    fuse_concat<<<cdiv((long long)NNODE * CATOUT, 256), 256, 0, stream>>>(
        fx1, x2, encA + (size_t)NEDGE * OUTDIM, alpha, xf_out,
        (long long)NNODE * CATOUT);

    // ---- 6. node-prob MLP --------------------------------------------------
    gemm(xf_out, np_W1, np_b1, h1, NNODE, 256, CATOUT, 2);   // elu
    gemm(h1, np_W2, np_b2, h2, NNODE, 128, 256, 2);          // elu
    np3_gemv<<<cdiv(NNODE, 256), 256, 0, stream>>>(h2, np_W3, np_b3, npb_out, NNODE);
}
